// MyCheckerboardRQS_72258529788184
// MI455X (gfx1250) — matrix operand order repaired and checked by reading, not compiled, not run on hardware
//
#include <hip/hip_runtime.h>
#include <hip/hip_bf16.h>

typedef __bf16 bf16_t;
typedef __attribute__((ext_vector_type(16))) __bf16 v16bf;
typedef __attribute__((ext_vector_type(8)))  __bf16 v8bf;
typedef __attribute__((ext_vector_type(8)))  float  v8f;
typedef int v4i_vs __attribute__((vector_size(4 * sizeof(int))));

#define TAIL 3.0f
#define MBW  0.001f
#define MBH  0.001f
#define MD_  0.001f

// --------------------------------------------------------------------------
// CDNA5 async global->LDS staging (ASYNCcnt-tracked), with safe fallback.
// --------------------------------------------------------------------------
#if defined(__AMDGCN__) && __has_builtin(__builtin_amdgcn_global_load_async_to_lds_b128)
#define HAVE_ASYNC_LDS 1
#else
#define HAVE_ASYNC_LDS 0
#endif

__device__ __forceinline__ void async_copy_b128(void* lds_dst, const void* gsrc) {
#if HAVE_ASYNC_LDS
  __builtin_amdgcn_global_load_async_to_lds_b128(
      (__attribute__((address_space(1))) v4i_vs*)const_cast<void*>(gsrc),
      (__attribute__((address_space(3))) v4i_vs*)lds_dst,
      0, 0);
#else
  *(uint4*)lds_dst = *(const uint4*)gsrc;
#endif
}

__device__ __forceinline__ void async_join() {
#if HAVE_ASYNC_LDS
#if __has_builtin(__builtin_amdgcn_s_wait_asynccnt)
  __builtin_amdgcn_s_wait_asynccnt(0);
#else
  asm volatile("s_wait_asynccnt 0" ::: "memory");
#endif
#endif
}

__device__ __forceinline__ float softplusf(float v) {
  return (v > 20.f) ? v : log1pf(__expf(v));
}

// ---------------------------------------------------------------------------
// Prep: x_frozen = checkerboard*x, concat conditioning, pad to 32ch, NHWC bf16
// ---------------------------------------------------------------------------
__global__ void prep_kernel(const float* __restrict__ x,
                            const float* __restrict__ cond,
                            bf16_t* __restrict__ inp,
                            int B, int H, int W) {
  int i = blockIdx.x * blockDim.x + threadIdx.x;
  int total = B * H * W;
  if (i >= total) return;
  int w = i % W;
  int h = (i / W) % H;
  int b = i / (W * H);
  int frozen = (h + w) & 1;                 // mask[h][w] = (h+w)%2
  bf16_t* dst = inp + (size_t)i * 32;
#pragma unroll
  for (int c = 0; c < 12; ++c) {
    float v = x[(((size_t)b * 12 + c) * H + h) * W + w];
    dst[c] = (bf16_t)(frozen ? v : 0.f);
  }
#pragma unroll
  for (int c = 0; c < 4; ++c) {
    float v = cond[(((size_t)b * 4 + c) * H + h) * W + w];
    dst[12 + c] = (bf16_t)v;
  }
#pragma unroll
  for (int c = 16; c < 32; ++c) dst[c] = (bf16_t)0.f;
}

// ---------------------------------------------------------------------------
// Weight pack: fp32 OIHW -> bf16 [tap=ky*3+kx][cout_pad][cin_pad]
// ---------------------------------------------------------------------------
__global__ void packw_kernel(const float* __restrict__ w, bf16_t* __restrict__ wp,
                             int O, int I, int NPAD, int CPAD) {
  int i = blockIdx.x * blockDim.x + threadIdx.x;
  int total = 9 * NPAD * CPAD;
  if (i >= total) return;
  int c = i % CPAD;
  int n = (i / CPAD) % NPAD;
  int t = i / (CPAD * NPAD);
  int ky = t / 3, kx = t % 3;
  float v = 0.f;
  if (n < O && c < I) v = w[(((size_t)n * I + c) * 3 + ky) * 3 + kx];
  wp[i] = (bf16_t)v;
}

__global__ void init_logdet_kernel(const float* __restrict__ ld,
                                   float* __restrict__ out, int B) {
  int i = threadIdx.x;
  if (i < B) out[i] = ld[i];
}

// ---------------------------------------------------------------------------
// Implicit-GEMM 3x3 SAME conv, NHWC bf16, Cout=128, fp32 accum via WMMA bf16.
// Block = one (b,h) image row: M=128(W) x N=128. 8 waves, each 16 M-rows.
// Double-buffered LDS: async DMA for step k+1 overlaps the WMMAs of step k;
// one barrier per k-step.
// ---------------------------------------------------------------------------
template <int CIN, bool RELU>
__global__ __launch_bounds__(256)
void conv_fwd(const bf16_t* __restrict__ in,   // [B,H,W,CIN]
              const bf16_t* __restrict__ wt,   // [9][128][CIN]
              const float*  __restrict__ bias, // [128]
              bf16_t* __restrict__ out,        // [B,H,W,128]
              int B, int H, int W) {
  constexpr int NT   = 128;
  constexpr int KC   = 32;
  constexpr int CCH  = CIN / KC;
  constexpr int KSTEPS = 9 * CCH;
  constexpr int AROW = KC + 8;   // bf16 elems per LDS row (pad vs bank conflicts)
  constexpr int BROW = KC + 8;
  __shared__ bf16_t As[2][128 * AROW];
  __shared__ bf16_t Bs[2][NT * BROW];

  const int bh = blockIdx.x;
  const int b = bh / H, h = bh % H;
  const int tid  = threadIdx.x;
  const int lane = tid & 31;
  const int wave = tid >> 5;          // M strip 0..7
  const int lm   = lane & 15;
  const int lk   = lane >> 4;         // 0/1

  v8f acc[NT / 16] = {};

  // Issue ASYNCcnt-tracked DMA for k-step `ks` into LDS buffer `buf`.
  auto stage = [&](int ks, int buf) {
    int tap = ks / CCH, cc = ks % CCH;
    int ky = tap / 3, kx = tap % 3;
    int hh = h + ky - 1;
    bool hok = (hh >= 0) && (hh < H);
    // A tile: 128 x 32 bf16 im2col row with zero halo
#pragma unroll
    for (int i = 0; i < 2; ++i) {
      int idx = tid + i * 256;
      int m = idx >> 2;               // w position 0..127
      int q = idx & 3;                // 16B quarter of 64B channel chunk
      int ww = m + kx - 1;
      void* ldst = &As[buf][m * AROW + q * 8];
      if (hok && ww >= 0 && ww < W) {
        const void* gsrc =
            in + (((size_t)(b * H + hh) * W + ww) * CIN + cc * KC) + q * 8;
        async_copy_b128(ldst, gsrc);
      } else {
        *(uint4*)ldst = make_uint4(0u, 0u, 0u, 0u);
      }
    }
    // B tile: 128 x 32 bf16 weights
    const bf16_t* wsrc = wt + ((size_t)tap * NT) * CIN + cc * KC;
#pragma unroll
    for (int i = 0; i < (NT * 4) / 256; ++i) {
      int idx = tid + i * 256;
      int n = idx >> 2;
      int q = idx & 3;
      async_copy_b128(&Bs[buf][n * BROW + q * 8], wsrc + (size_t)n * CIN + q * 8);
    }
  };

  stage(0, 0);
  async_join();
  __syncthreads();

  for (int ks = 0; ks < KSTEPS; ++ks) {
    const int cur = ks & 1;
    if (ks + 1 < KSTEPS) stage(ks + 1, cur ^ 1);   // DMA next step under WMMAs
    // ---- A fragment: lane<16: M=lm,K{0..7,16..23}; lane>=16: K{8..15,24..31}
    v8bf a0 = *(const v8bf*)&As[cur][(wave * 16 + lm) * AROW + lk * 8];
    v8bf a1 = *(const v8bf*)&As[cur][(wave * 16 + lm) * AROW + 16 + lk * 8];
    v16bf a = __builtin_shufflevector(a0, a1, 0, 1, 2, 3, 4, 5, 6, 7,
                                      8, 9, 10, 11, 12, 13, 14, 15);
    v8bf b0 = *(const v8bf*)&Bs[cur][lm * BROW + lk * 8];
    v8bf b1 = *(const v8bf*)&Bs[cur][lm * BROW + 16 + lk * 8];
#pragma unroll
    for (int nt = 0; nt < NT / 16; ++nt) {
      v16bf bb = __builtin_shufflevector(b0, b1, 0, 1, 2, 3, 4, 5, 6, 7,
                                         8, 9, 10, 11, 12, 13, 14, 15);
      if (nt + 1 < NT / 16) {
        b0 = *(const v8bf*)&Bs[cur][((nt + 1) * 16 + lm) * BROW + lk * 8];
        b1 = *(const v8bf*)&Bs[cur][((nt + 1) * 16 + lm) * BROW + 16 + lk * 8];
      }
      acc[nt] = __builtin_amdgcn_wmma_f32_16x16x32_bf16(
          false, a, false, bb, (short)0, acc[nt], false, false);
    }
    async_join();       // next step's DMA landed (own)
    __syncthreads();    // all waves: done reading cur, next buffer fully written
  }

  // ---- epilogue: +bias, relu, bf16 NHWC store ----
#pragma unroll
  for (int nt = 0; nt < NT / 16; ++nt) {
    int n = nt * 16 + lm;
    float bz = bias[n];
#pragma unroll
    for (int e = 0; e < 8; ++e) {
      int m = wave * 16 + e + 8 * lk;
      float v = acc[nt][e] + bz;
      if (RELU) v = v > 0.f ? v : 0.f;
      out[(((size_t)(b * H + h) * W + m) * NT) + n] = (bf16_t)v;
    }
  }
}

// ---------------------------------------------------------------------------
// Conv3 (Cout 276 -> padded 288) fused with RQS spline + logdet reduction.
// Block = one (b,h) row, M=128 x N=288; 8 waves x (16M x 18 N-tiles).
// (Single-buffered: 18 WMMAs/step already cover load latency; LDS budget.)
// ---------------------------------------------------------------------------
__global__ __launch_bounds__(256)
void conv3_rqs(const bf16_t* __restrict__ in,   // [B,H,W,128]
               const bf16_t* __restrict__ wt,   // [9][288][128]
               const float*  __restrict__ bias, // [276]
               const float*  __restrict__ x,    // [B,12,H,W]
               float* __restrict__ xout,        // [B,12,H,W]
               float* __restrict__ ldout,       // [B]
               int B, int H, int W) {
  constexpr int CIN = 128, NT = 288, KC = 32, CCH = 4;
  constexpr int AROW = KC + 8, BROW = KC + 8, PROW = 292;
  __shared__ bf16_t As[128 * AROW];
  __shared__ bf16_t Bs[NT * BROW];
  __shared__ float  Ps[16 * PROW];
  __shared__ float  wred[8];

  const int bh = blockIdx.x;
  const int b = bh / H, h = bh % H;
  const int tid  = threadIdx.x;
  const int lane = tid & 31;
  const int wave = tid >> 5;
  const int lm   = lane & 15;
  const int lk   = lane >> 4;

  v8f acc[NT / 16] = {};

  for (int ky = 0; ky < 3; ++ky) {
    const int hh = h + ky - 1;
    const bool hok = (hh >= 0) && (hh < H);
    for (int kx = 0; kx < 3; ++kx) {
      const bf16_t* wtap = wt + ((size_t)(ky * 3 + kx) * NT) * CIN;
      if (ky * 3 + kx < 8) __builtin_prefetch(wtap + (size_t)NT * CIN, 0, 1);
      for (int cc = 0; cc < CCH; ++cc) {
#pragma unroll
        for (int i = 0; i < 2; ++i) {
          int idx = tid + i * 256;
          int m = idx >> 2;
          int q = idx & 3;
          int ww = m + kx - 1;
          void* ldst = &As[m * AROW + q * 8];
          if (hok && ww >= 0 && ww < W) {
            const void* gsrc =
                in + (((size_t)(b * H + hh) * W + ww) * CIN + cc * KC) + q * 8;
            async_copy_b128(ldst, gsrc);
          } else {
            *(uint4*)ldst = make_uint4(0u, 0u, 0u, 0u);
          }
        }
        const bf16_t* wsrc = wtap + (size_t)cc * KC;
#pragma unroll
        for (int i = 0; i < 5; ++i) {
          int idx = tid + i * 256;
          if (idx < NT * 4) {
            int n = idx >> 2;
            int q = idx & 3;
            async_copy_b128(&Bs[n * BROW + q * 8], wsrc + (size_t)n * CIN + q * 8);
          }
        }
        async_join();
        __syncthreads();
        v8bf a0 = *(const v8bf*)&As[(wave * 16 + lm) * AROW + lk * 8];
        v8bf a1 = *(const v8bf*)&As[(wave * 16 + lm) * AROW + 16 + lk * 8];
        v16bf a = __builtin_shufflevector(a0, a1, 0, 1, 2, 3, 4, 5, 6, 7,
                                          8, 9, 10, 11, 12, 13, 14, 15);
        v8bf b0 = *(const v8bf*)&Bs[lm * BROW + lk * 8];
        v8bf b1 = *(const v8bf*)&Bs[lm * BROW + 16 + lk * 8];
#pragma unroll
        for (int nt = 0; nt < NT / 16; ++nt) {
          v16bf bb = __builtin_shufflevector(b0, b1, 0, 1, 2, 3, 4, 5, 6, 7,
                                             8, 9, 10, 11, 12, 13, 14, 15);
          if (nt + 1 < NT / 16) {
            b0 = *(const v8bf*)&Bs[((nt + 1) * 16 + lm) * BROW + lk * 8];
            b1 = *(const v8bf*)&Bs[((nt + 1) * 16 + lm) * BROW + 16 + lk * 8];
          }
          acc[nt] = __builtin_amdgcn_wmma_f32_16x16x32_bf16(
              false, a, false, bb, (short)0, acc[nt], false, false);
        }
        __syncthreads();
      }
    }
  }

  // ---- fused RQS epilogue: 8 phases, one wave's 16x288 strip at a time ----
  float mylad = 0.f;
  for (int ph = 0; ph < 8; ++ph) {
    __syncthreads();
    if (wave == ph) {
#pragma unroll
      for (int nt = 0; nt < NT / 16; ++nt) {
        int n = nt * 16 + lm;
        float bz = (n < 276) ? bias[n] : 0.f;
#pragma unroll
        for (int e = 0; e < 8; ++e) {
          int r = e + 8 * lk;                       // row within strip
          Ps[r * PROW + n] = acc[nt][e] + bz;
        }
      }
    }
    __syncthreads();
    if (tid < 192) {                                // 16 w-positions x 12 channels
      int ml = tid / 12;
      int c  = tid % 12;
      int w  = ph * 16 + ml;
      const float* P = &Ps[ml * PROW + c * 23];
      float xv = x[(((size_t)b * 12 + c) * H + h) * W + w];

      // widths: softmax -> knots cw[0..8]
      float u[8], ex[8];
      float mx, s, run, invs;
      float cw[9], ch[9], dk[9];
#pragma unroll
      for (int i = 0; i < 8; ++i) u[i] = P[i];
      mx = u[0];
#pragma unroll
      for (int i = 1; i < 8; ++i) mx = fmaxf(mx, u[i]);
      s = 0.f;
#pragma unroll
      for (int i = 0; i < 8; ++i) { ex[i] = __expf(u[i] - mx); s += ex[i]; }
      invs = 1.f / s;
      cw[0] = -TAIL; run = 0.f;
#pragma unroll
      for (int i = 0; i < 8; ++i) {
        run += MBW + (1.f - MBW * 8.f) * ex[i] * invs;
        cw[i + 1] = 2.f * TAIL * run - TAIL;
      }
      cw[8] = TAIL;
      // heights
#pragma unroll
      for (int i = 0; i < 8; ++i) u[i] = P[8 + i];
      mx = u[0];
#pragma unroll
      for (int i = 1; i < 8; ++i) mx = fmaxf(mx, u[i]);
      s = 0.f;
#pragma unroll
      for (int i = 0; i < 8; ++i) { ex[i] = __expf(u[i] - mx); s += ex[i]; }
      invs = 1.f / s;
      ch[0] = -TAIL; run = 0.f;
#pragma unroll
      for (int i = 0; i < 8; ++i) {
        run += MBH + (1.f - MBH * 8.f) * ex[i] * invs;
        ch[i + 1] = 2.f * TAIL * run - TAIL;
      }
      ch[8] = TAIL;
      // derivatives (padded ends -> exactly 1.0)
      dk[0] = 1.f; dk[8] = 1.f;
#pragma unroll
      for (int i = 0; i < 7; ++i) dk[i + 1] = MD_ + softplusf(P[16 + i]);

      float xc = fminf(TAIL, fmaxf(-TAIL, xv));
      int idx = 0;
#pragma unroll
      for (int i = 1; i < 9; ++i) idx += (xc >= cw[i]) ? 1 : 0;
      if (idx > 7) idx = 7;

      float icw = cw[idx], iw = cw[idx + 1] - cw[idx];
      float ich = ch[idx], ihh = ch[idx + 1] - ch[idx];
      float dl = dk[idx], dr = dk[idx + 1];
      float delta = ihh / iw;
      float th  = (xc - icw) / iw;
      float th1 = 1.f - th;
      float tt  = th * th1;
      float num = ihh * (delta * th * th + dl * tt);
      float den = delta + (dl + dr - 2.f * delta) * tt;
      float y   = ich + num / den;
      float dnum = delta * delta * (dr * th * th + 2.f * delta * tt + dl * th1 * th1);
      float lad  = __logf(dnum) - 2.f * __logf(den);
      bool inside = (xv >= -TAIL) && (xv <= TAIL);
      float yo = inside ? y : xv;
      if (!inside) lad = 0.f;
      int frozen = (h + w) & 1;
      xout[(((size_t)b * 12 + c) * H + h) * W + w] = frozen ? xv : yo;
      if (!frozen) mylad += lad;
    }
  }

  // ---- block reduce logdet, one atomic per block ----
#pragma unroll
  for (int off = 16; off > 0; off >>= 1) mylad += __shfl_down(mylad, off);
  if (lane == 0) wred[wave] = mylad;
  __syncthreads();
  if (tid == 0) {
    float t = 0.f;
#pragma unroll
    for (int i = 0; i < 8; ++i) t += wred[i];
    atomicAdd(&ldout[b], t);
  }
}

// ---------------------------------------------------------------------------
extern "C" void kernel_launch(void* const* d_in, const int* in_sizes, int n_in,
                              void* d_out, int out_size, void* d_ws, size_t ws_size,
                              hipStream_t stream) {
  (void)in_sizes; (void)n_in; (void)out_size; (void)ws_size;
  const int B = 16, C = 12, H = 128, W = 128;

  const float* x    = (const float*)d_in[0];
  const float* ld   = (const float*)d_in[1];
  const float* cond = (const float*)d_in[2];
  const float* w1   = (const float*)d_in[3];
  const float* b1   = (const float*)d_in[4];
  const float* w2   = (const float*)d_in[5];
  const float* b2   = (const float*)d_in[6];
  const float* w3   = (const float*)d_in[7];
  const float* b3   = (const float*)d_in[8];

  float* xout  = (float*)d_out;
  float* ldout = xout + (size_t)B * C * H * W;

  char* ws = (char*)d_ws;
  bf16_t* inp = (bf16_t*)ws; ws += (size_t)B * H * W * 32  * sizeof(bf16_t);
  bf16_t* a1  = (bf16_t*)ws; ws += (size_t)B * H * W * 128 * sizeof(bf16_t);
  bf16_t* a2  = (bf16_t*)ws; ws += (size_t)B * H * W * 128 * sizeof(bf16_t);
  bf16_t* wp1 = (bf16_t*)ws; ws += (size_t)9 * 128 * 32  * sizeof(bf16_t);
  bf16_t* wp2 = (bf16_t*)ws; ws += (size_t)9 * 128 * 128 * sizeof(bf16_t);
  bf16_t* wp3 = (bf16_t*)ws; ws += (size_t)9 * 288 * 128 * sizeof(bf16_t);

  int npix = B * H * W;
  prep_kernel<<<(npix + 255) / 256, 256, 0, stream>>>(x, cond, inp, B, H, W);
  packw_kernel<<<(9 * 128 * 32  + 255) / 256, 256, 0, stream>>>(w1, wp1, 128, 16,  128, 32);
  packw_kernel<<<(9 * 128 * 128 + 255) / 256, 256, 0, stream>>>(w2, wp2, 128, 128, 128, 128);
  packw_kernel<<<(9 * 288 * 128 + 255) / 256, 256, 0, stream>>>(w3, wp3, 276, 128, 288, 128);

  conv_fwd<32,  true><<<B * H, 256, 0, stream>>>(inp, wp1, b1, a1, B, H, W);
  conv_fwd<128, true><<<B * H, 256, 0, stream>>>(a1,  wp2, b2, a2, B, H, W);

  init_logdet_kernel<<<1, 32, 0, stream>>>(ld, ldout, B);
  conv3_rqs<<<B * H, 256, 0, stream>>>(a2, wp3, b3, x, xout, ldout, B, H, W);
}
